// SearchTransfer2_49452253446913
// MI455X (gfx1250) — matrix operand order repaired and checked by reading, not compiled, not run on hardware
//
#include <hip/hip_runtime.h>
#include <hip/hip_bf16.h>

// ---------------------------------------------------------------------------
// SearchTransfer pipeline for MI455X (gfx1250, wave32).
// All GEMM-shaped work is done with V_WMMA_F32_16X16X32_BF16 (fp32 accum).
// Fragment layouts per CDNA5 ISA 7.12.2 (16-bit A 16x32, B 32x16, f32 C/D).
// ---------------------------------------------------------------------------

typedef __bf16 bf16_t;
typedef __attribute__((ext_vector_type(16))) __bf16 bf16x16;
typedef __attribute__((ext_vector_type(8)))  float  f32x8;

__device__ __forceinline__ float gelu_exact(float x) {
  return 0.5f * x * (1.0f + erff(x * 0.70710678118654752440f));
}

__device__ __forceinline__ f32x8 wmma_bf16(bf16x16 a, bf16x16 b, f32x8 c) {
  return __builtin_amdgcn_wmma_f32_16x16x32_bf16(false, a, false, b,
                                                 (short)0, c, false, false);
}

// A fragment (16x32 bf16) from LDS row-major tile, row length >= 32.
// lane<16: row M=lane, K {0..7, 16..23};  lane>=16: row M=lane-16, K {8..15, 24..31}.
// p = &tile[row*ldk] (start of the 32-wide K chunk), k0 = (lane&16)?8:0.
__device__ __forceinline__ bf16x16 frag_a(const bf16_t* p, int k0) {
  union { uint4 v; bf16_t h[8]; } u0, u1;
  u0.v = *(const uint4*)(p + k0);
  u1.v = *(const uint4*)(p + k0 + 16);
  bf16x16 a;
#pragma unroll
  for (int i = 0; i < 8; ++i) { a[i] = u0.h[i]; a[8 + i] = u1.h[i]; }
  return a;
}

// B fragment (32x16 bf16) from LDS laid out [N][K].
// lane<16: col N=lane, K {0..7, 16..23};  lane>=16: col N=lane-16, K {8..15, 24..31}.
// p = &tile[n*ldk + kbase], kbase = (lane&16)?8:0.
__device__ __forceinline__ bf16x16 frag_b(const bf16_t* p) {
  union { uint4 v[2]; bf16_t h[16]; } u;
  u.v[0] = *(const uint4*)p;
  u.v[1] = *(const uint4*)(p + 16);
  bf16x16 b;
#pragma unroll
  for (int i = 0; i < 16; ++i) b[i] = u.h[i];
  return b;
}

// ---------------------------------------------------------------------------
// float -> bf16 cast (row-major flatten already matches K-order for wq/wk/W1/W2)
// ---------------------------------------------------------------------------
__global__ __launch_bounds__(256) void cast_bf16_k(const float* __restrict__ x,
                                                   bf16_t* __restrict__ y,
                                                   long long n) {
  long long i = (long long)blockIdx.x * 256 + threadIdx.x;
  long long st = (long long)gridDim.x * 256;
  for (; i < n; i += st) y[i] = (bf16_t)x[i];
}

// ---------------------------------------------------------------------------
// Conv weight repack to tap-major K order: Wp[o][t*64 + c] = w[o][c][t/3][t%3].
// Tap-major K lets the im2col gather decompose k with shifts only (t uniform
// per 32-wide K chunk), killing the div-by-9 VALU storm in the conv kernel.
// ---------------------------------------------------------------------------
__global__ __launch_bounds__(256) void pack_conv_w_k(
    const float* __restrict__ w,  // [64][64][3][3]
    bf16_t* __restrict__ Wp) {    // [64][576]
  const int o = blockIdx.x;
  for (int k = threadIdx.x; k < 576; k += 256) {
    int t = k >> 6, c = k & 63;
    Wp[o * 576 + k] = (bf16_t)w[(o * 64 + c) * 9 + t];
  }
}

// ---------------------------------------------------------------------------
// Generic bf16 WMMA GEMM: C[M,N] = A[M,K] @ B (+bias)
//   B_NK=true : B source is [N][K] row-major (weights / kp for Q·K^T)
//   B_NK=false: B source is [K][N] row-major (V for attn·V) -> transposed stage
//   EPI: 0 = f32 out + bias[N], 1 = f32 out, 2 = bf16 out
// Block 256 thr (8 waves), tile 128x128x32, wave tile 64x32 (4x2 frags).
// Next-chunk A/B prefetch (global_prefetch_b8) overlaps HBM fill with WMMAs.
// ---------------------------------------------------------------------------
template <bool B_NK, int EPI>
__global__ __launch_bounds__(256) void gemm_bf16_k(
    const bf16_t* __restrict__ A, const bf16_t* __restrict__ B,
    void* __restrict__ C, const float* __restrict__ bias,
    int M, int N, int K, long long sA, long long sB, long long sC) {
  constexpr int BM = 128, BN = 128, BK = 32, BKP = 40;  // padded LDS row
  __shared__ bf16_t As[BM * BKP];
  __shared__ bf16_t Bs[BN * BKP];
  const int z = blockIdx.z;
  const bf16_t* Ab = A + (long long)z * sA;
  const bf16_t* Bb = B + (long long)z * sB;
  const int tid = threadIdx.x, lane = tid & 31, wave = tid >> 5;
  const int wm = wave >> 2, wn = wave & 3;  // 2 x 4 wave grid
  const int M0 = blockIdx.y * BM, N0 = blockIdx.x * BN;
  const int ar = lane & 15;
  const int ak0 = (lane & 16) ? 8 : 0;
  const int bk0 = (lane & 16) ? 8 : 0;

  f32x8 acc[4][2];
#pragma unroll
  for (int i = 0; i < 4; ++i)
#pragma unroll
    for (int j = 0; j < 2; ++j)
#pragma unroll
      for (int g = 0; g < 8; ++g) acc[i][j][g] = 0.0f;

  for (int K0 = 0; K0 < K; K0 += BK) {
#pragma unroll
    for (int it = 0; it < 4; ++it) {           // A: 128x32, 8B vectors
      int idx = tid + it * 256;
      int r = idx >> 3, c4 = (idx & 7) * 4;
      *(uint2*)&As[r * BKP + c4] =
          *(const uint2*)&Ab[(long long)(M0 + r) * K + K0 + c4];
    }
    if (B_NK) {
#pragma unroll
      for (int it = 0; it < 4; ++it) {         // B: 128x32, 8B vectors
        int idx = tid + it * 256;
        int r = idx >> 3, c4 = (idx & 7) * 4;
        *(uint2*)&Bs[r * BKP + c4] =
            *(const uint2*)&Bb[(long long)(N0 + r) * K + K0 + c4];
      }
    } else {
#pragma unroll
      for (int it = 0; it < 16; ++it) {        // transpose-stage [K][N] -> [N][K]
        int idx = tid + it * 256;
        int kk = idx >> 7, n = idx & 127;
        Bs[n * BKP + kk] = Bb[(long long)(K0 + kk) * N + N0 + n];
      }
    }
    // Prefetch next K chunk into cache while this chunk's WMMAs run.
    if (K0 + BK < K) {
      int pr = tid >> 1, pc = (tid & 1) * 16;  // 256 thr -> 128 rows x 2 halves
      __builtin_prefetch(&Ab[(long long)(M0 + pr) * K + K0 + BK + pc], 0, 0);
      if (B_NK) {
        __builtin_prefetch(&Bb[(long long)(N0 + pr) * K + K0 + BK + pc], 0, 0);
      } else {
        __builtin_prefetch(
            &Bb[(long long)(K0 + BK + (tid >> 3)) * N + N0 + (tid & 7) * 16], 0,
            0);
      }
    }
    __syncthreads();
    bf16x16 af[4], bfr[2];
#pragma unroll
    for (int i = 0; i < 4; ++i)
      af[i] = frag_a(&As[(wm * 64 + i * 16 + ar) * BKP], ak0);
#pragma unroll
    for (int j = 0; j < 2; ++j)
      bfr[j] = frag_b(&Bs[(wn * 32 + j * 16 + ar) * BKP + bk0]);
#pragma unroll
    for (int i = 0; i < 4; ++i)
#pragma unroll
      for (int j = 0; j < 2; ++j) acc[i][j] = wmma_bf16(af[i], bfr[j], acc[i][j]);
    __syncthreads();
  }

  const int cn = lane & 15, cm = (lane >> 4) * 8;
#pragma unroll
  for (int i = 0; i < 4; ++i)
#pragma unroll
    for (int j = 0; j < 2; ++j) {
      int col = N0 + wn * 32 + j * 16 + cn;
#pragma unroll
      for (int g = 0; g < 8; ++g) {
        int row = M0 + wm * 64 + i * 16 + cm + g;
        long long o = (long long)z * sC + (long long)row * N + col;
        float v = acc[i][j][g];
        if (EPI == 0)      ((float*)C)[o] = v + bias[col];
        else if (EPI == 1) ((float*)C)[o] = v;
        else               ((bf16_t*)C)[o] = (bf16_t)v;
      }
    }
}

// ---------------------------------------------------------------------------
// 3x3 conv (SAME, C=64->64) as implicit GEMM + GELU, writing directly into
// the unfolded layout [B][L=1024][4096], d = c*64 + (y%8)*8 + (x%8).
// K order is tap-major: k = t*64 + c (t = 3x3 tap, c = input channel), so a
// 32-wide K chunk has uniform tap -> shift/mask-only index math in the gather.
// grid: (HW/128, B, 3 convs). K = 9*64 = 576.
// ---------------------------------------------------------------------------
__global__ __launch_bounds__(256) void conv3x3_gelu_unfold_k(
    const bf16_t* __restrict__ featbf,  // [B][64][256][256]
    const bf16_t* __restrict__ Wc,      // [3][64][576] tap-major packs
    const float* __restrict__ bias0, const float* __restrict__ bias1,
    const float* __restrict__ bias2,
    bf16_t* __restrict__ out0, bf16_t* __restrict__ out1,
    bf16_t* __restrict__ out2) {
  constexpr int BM = 128, BK = 32, BKP = 40, KTOT = 576;
  __shared__ bf16_t As[BM * BKP];
  __shared__ bf16_t Bs[64 * BKP];
  const int b = blockIdx.y, s = blockIdx.z;
  const bf16_t* Ws = Wc + (size_t)s * (64 * KTOT);
  const float* bias = (s == 0) ? bias0 : (s == 1 ? bias1 : bias2);
  bf16_t* out = (s == 0) ? out0 : (s == 1 ? out1 : out2);
  const bf16_t* fb = featbf + (size_t)b * 64 * 256 * 256;

  const int tid = threadIdx.x, lane = tid & 31, wave = tid >> 5;
  const int p0 = blockIdx.x * BM;
  const int ar = lane & 15;
  const int ak0 = (lane & 16) ? 8 : 0;
  const int bk0 = (lane & 16) ? 8 : 0;
  const int kk = tid & 31;        // K-within-chunk, fixed per thread
  const int r0 = tid >> 5;        // base tile row, fixed per thread

  f32x8 acc[4];
#pragma unroll
  for (int j = 0; j < 4; ++j)
#pragma unroll
    for (int g = 0; g < 8; ++g) acc[j][g] = 0.0f;

  for (int K0 = 0; K0 < KTOT; K0 += BK) {
    const int t = K0 >> 6;                    // uniform tap for this chunk
    const int dy = t / 3 - 1, dx = t % 3 - 1; // SALU (uniform)
    const int c = (K0 & 63) + kk;             // channel, fixed per thread
    const bf16_t* fc = fb + ((size_t)c << 16);
#pragma unroll
    for (int it = 0; it < 16; ++it) {         // im2col gather 128x32
      int r = r0 + it * 8;
      int p = p0 + r, y = p >> 8, x = p & 255;
      int yy = y + dy, xx = x + dx;
      bf16_t bv = (bf16_t)0.0f;
      if ((unsigned)yy < 256u && (unsigned)xx < 256u)
        bv = fc[(yy << 8) + xx];
      As[r * BKP + kk] = bv;
    }
#pragma unroll
    for (int it = 0; it < 2; ++it) {          // weights: 64x32 chunk
      int idx = tid + it * 256;
      int rr = idx >> 3, c4 = (idx & 7) * 4;
      *(uint2*)&Bs[rr * BKP + c4] = *(const uint2*)&Ws[rr * KTOT + K0 + c4];
    }
    __syncthreads();
    bf16x16 af = frag_a(&As[(wave * 16 + ar) * BKP], ak0);
#pragma unroll
    for (int j = 0; j < 4; ++j) {
      bf16x16 bfr = frag_b(&Bs[(j * 16 + ar) * BKP + bk0]);
      acc[j] = wmma_bf16(af, bfr, acc[j]);
    }
    __syncthreads();
  }

  const int cn = lane & 15, cm = (lane >> 4) * 8;
  bf16_t* ob = out + (size_t)b * 1024 * 4096;
#pragma unroll
  for (int j = 0; j < 4; ++j) {
    int n = j * 16 + cn;
#pragma unroll
    for (int g = 0; g < 8; ++g) {
      int p = p0 + wave * 16 + cm + g;
      int y = p >> 8, x = p & 255;
      int l = (y >> 3) * 32 + (x >> 3);
      int d = n * 64 + (y & 7) * 8 + (x & 7);
      ob[(size_t)l * 4096 + d] = (bf16_t)gelu_exact(acc[j][g] + bias[n]);
    }
  }
}

// ---------------------------------------------------------------------------
// BatchNorm1d batch stats over (B,L)=4096 samples per channel.
// Emits fused affine (a, b): y = x*a + b, pre-multiplied by `postscale`
// (1/sqrt(32) on each of q,k so scores carry the 1/32 attention scale).
// ---------------------------------------------------------------------------
__global__ __launch_bounds__(256) void bn_stats_k(
    const float* __restrict__ X,  // [4096][1024]
    const float* __restrict__ gamma, const float* __restrict__ beta,
    float* __restrict__ sa, float* __restrict__ sb, float postscale) {
  __shared__ float r1[256], r2[256];
  const int ch = blockIdx.x, tid = threadIdx.x;
  float s = 0.f, s2 = 0.f;
  for (int i = tid; i < 4096; i += 256) {
    float v = X[(long long)i * 1024 + ch];
    s += v; s2 += v * v;
  }
  r1[tid] = s; r2[tid] = s2;
  __syncthreads();
  for (int st = 128; st > 0; st >>= 1) {
    if (tid < st) { r1[tid] += r1[tid + st]; r2[tid] += r2[tid + st]; }
    __syncthreads();
  }
  if (tid == 0) {
    float m = r1[0] * (1.0f / 4096.0f);
    float var = r2[0] * (1.0f / 4096.0f) - m * m;
    float rs = rsqrtf(var + 1e-5f);
    float a = gamma[ch] * rs;
    sa[ch] = a * postscale;
    sb[ch] = (beta[ch] - m * a) * postscale;
  }
}

__global__ __launch_bounds__(256) void bn_apply_k(
    const float* __restrict__ X, const float* __restrict__ sa,
    const float* __restrict__ sb, bf16_t* __restrict__ Y, long long n) {
  long long i = (long long)blockIdx.x * 256 + threadIdx.x;
  long long st = (long long)gridDim.x * 256;
  for (; i < n; i += st) {
    int ch = (int)(i & 1023);
    Y[i] = (bf16_t)(X[i] * sa[ch] + sb[ch]);
  }
}

// ---------------------------------------------------------------------------
// Row softmax over 1024 cols, fp32 in -> bf16 out. grid = B*1024 rows.
// ---------------------------------------------------------------------------
__global__ __launch_bounds__(256) void softmax_row_k(
    const float* __restrict__ S, bf16_t* __restrict__ A) {
  __shared__ float red[256];
  const long long row = blockIdx.x;
  const float* r = S + row * 1024;
  const int tid = threadIdx.x;
  float v0 = r[tid], v1 = r[tid + 256], v2 = r[tid + 512], v3 = r[tid + 768];
  float m = fmaxf(fmaxf(v0, v1), fmaxf(v2, v3));
  red[tid] = m;
  __syncthreads();
  for (int st = 128; st > 0; st >>= 1) {
    if (tid < st) red[tid] = fmaxf(red[tid], red[tid + st]);
    __syncthreads();
  }
  m = red[0];
  __syncthreads();
  float e0 = __expf(v0 - m), e1 = __expf(v1 - m);
  float e2 = __expf(v2 - m), e3 = __expf(v3 - m);
  red[tid] = e0 + e1 + e2 + e3;
  __syncthreads();
  for (int st = 128; st > 0; st >>= 1) {
    if (tid < st) red[tid] += red[tid + st];
    __syncthreads();
  }
  float inv = 1.0f / red[0];
  bf16_t* o = A + row * 1024;
  o[tid] = (bf16_t)(e0 * inv);
  o[tid + 256] = (bf16_t)(e1 * inv);
  o[tid + 512] = (bf16_t)(e2 * inv);
  o[tid + 768] = (bf16_t)(e3 * inv);
}

// ---------------------------------------------------------------------------
// Fused fold + FFN (1x1 convs 64->256->64, GELU) + residual.
// Reads attn output directly from unfolded layout (fold is re-indexing).
// Per block: 64 pixels. Phase1 hid = gelu(X@W1^T+b1) kept in LDS (bf16);
// Phase2 y = gelu(hid@W2^T+b2) + feat -> d_out (fp32).
// ---------------------------------------------------------------------------
__global__ __launch_bounds__(256) void ffn_fused_k(
    const bf16_t* __restrict__ out_un,  // [B][1024][4096]
    const bf16_t* __restrict__ W1,      // [256][64]
    const float* __restrict__ b1,       // [256]
    const bf16_t* __restrict__ W2,      // [64][256]
    const float* __restrict__ b2,       // [64]
    const float* __restrict__ feat,     // [B][64][256][256]
    float* __restrict__ out) {
  constexpr int BKP = 40, HLD = 264;  // padded LDS rows
  __shared__ bf16_t Xs[64 * BKP];
  __shared__ bf16_t Bs[256 * BKP];
  __shared__ bf16_t Hs[64 * HLD];
  const int b = blockIdx.y;
  const int p0 = blockIdx.x * 64;
  const int tid = threadIdx.x, lane = tid & 31, wave = tid >> 5;
  const int ar = lane & 15;
  const int ak0 = (lane & 16) ? 8 : 0;
  const int bk0 = (lane & 16) ? 8 : 0;
  const int cn = lane & 15, cm = (lane >> 4) * 8;
  const bf16_t* xb = out_un + (size_t)b * 1024 * 4096;

  {  // -------- phase 1: hid[64][256] --------
    const int wm = wave >> 2, wn = wave & 3;  // 2 x 4
    f32x8 acc[2][4];
#pragma unroll
    for (int i = 0; i < 2; ++i)
#pragma unroll
      for (int j = 0; j < 4; ++j)
#pragma unroll
        for (int g = 0; g < 8; ++g) acc[i][j][g] = 0.0f;
    for (int K0 = 0; K0 < 64; K0 += 32) {
#pragma unroll
      for (int it = 0; it < 8; ++it) {  // gather X from unfolded layout
        int idx = tid + it * 256;
        int r = idx >> 5, kk = idx & 31;
        int p = p0 + r, y = p >> 8, x = p & 255;
        int c = K0 + kk;
        int l = (y >> 3) * 32 + (x >> 3);
        int d = c * 64 + (y & 7) * 8 + (x & 7);
        Xs[r * BKP + kk] = xb[(size_t)l * 4096 + d];
      }
#pragma unroll
      for (int it = 0; it < 8; ++it) {  // W1 chunk [256][32]
        int idx = tid + it * 256;
        int r = idx >> 3, c4 = (idx & 7) * 4;
        *(uint2*)&Bs[r * BKP + c4] = *(const uint2*)&W1[r * 64 + K0 + c4];
      }
      __syncthreads();
      bf16x16 af[2];
#pragma unroll
      for (int i = 0; i < 2; ++i)
        af[i] = frag_a(&Xs[(wm * 32 + i * 16 + ar) * BKP], ak0);
#pragma unroll
      for (int j = 0; j < 4; ++j) {
        bf16x16 bfr = frag_b(&Bs[(wn * 64 + j * 16 + ar) * BKP + bk0]);
#pragma unroll
        for (int i = 0; i < 2; ++i) acc[i][j] = wmma_bf16(af[i], bfr, acc[i][j]);
      }
      __syncthreads();
    }
#pragma unroll
    for (int i = 0; i < 2; ++i)
#pragma unroll
      for (int j = 0; j < 4; ++j) {
        int n = wn * 64 + j * 16 + cn;
#pragma unroll
        for (int g = 0; g < 8; ++g) {
          int m = wm * 32 + i * 16 + cm + g;
          Hs[m * HLD + n] = (bf16_t)gelu_exact(acc[i][j][g] + b1[n]);
        }
      }
    __syncthreads();
  }

  {  // -------- phase 2: y = gelu(H @ W2^T + b2) + feat --------
    const int wm = wave >> 1, wn = wave & 1;  // 4 x 2
    f32x8 acc[2];
#pragma unroll
    for (int j = 0; j < 2; ++j)
#pragma unroll
      for (int g = 0; g < 8; ++g) acc[j][g] = 0.0f;
    for (int K0 = 0; K0 < 256; K0 += 32) {
#pragma unroll
      for (int it = 0; it < 2; ++it) {  // W2 chunk [64][32]
        int idx = tid + it * 256;
        int r = idx >> 3, c4 = (idx & 7) * 4;
        *(uint2*)&Bs[r * BKP + c4] = *(const uint2*)&W2[r * 256 + K0 + c4];
      }
      __syncthreads();
      bf16x16 af = frag_a(&Hs[(wm * 16 + ar) * HLD + K0], ak0);
#pragma unroll
      for (int j = 0; j < 2; ++j) {
        bf16x16 bfr = frag_b(&Bs[(wn * 32 + j * 16 + ar) * BKP + bk0]);
        acc[j] = wmma_bf16(af, bfr, acc[j]);
      }
      __syncthreads();
    }
#pragma unroll
    for (int j = 0; j < 2; ++j) {
      int n = wn * 32 + j * 16 + cn;
#pragma unroll
      for (int g = 0; g < 8; ++g) {
        int m = wm * 16 + cm + g;
        int p = p0 + m, y = p >> 8, x = p & 255;
        size_t o = ((size_t)(b * 64 + n) * 256 + y) * 256 + x;
        out[o] = gelu_exact(acc[j][g] + b2[n]) + feat[o];
      }
    }
  }
}

// ---------------------------------------------------------------------------
// Host orchestration
// ---------------------------------------------------------------------------
extern "C" void kernel_launch(void* const* d_in, const int* in_sizes, int n_in,
                              void* d_out, int out_size, void* d_ws,
                              size_t ws_size, hipStream_t stream) {
  (void)in_sizes; (void)n_in; (void)out_size; (void)ws_size;
  const float* feat  = (const float*)d_in[0];
  const float* w_enc = (const float*)d_in[1];
  const float* b_enc = (const float*)d_in[2];
  const float* w_kc  = (const float*)d_in[3];
  const float* b_kc  = (const float*)d_in[4];
  const float* w_vc  = (const float*)d_in[5];
  const float* b_vc  = (const float*)d_in[6];
  const float* wq    = (const float*)d_in[7];
  const float* bq    = (const float*)d_in[8];
  const float* wk    = (const float*)d_in[9];
  const float* bk    = (const float*)d_in[10];
  const float* gq    = (const float*)d_in[11];
  const float* betaq = (const float*)d_in[12];
  const float* gk    = (const float*)d_in[13];
  const float* betak = (const float*)d_in[14];
  const float* w_f1  = (const float*)d_in[15];
  const float* b_f1  = (const float*)d_in[16];
  const float* w_f2  = (const float*)d_in[17];
  const float* b_f2  = (const float*)d_in[18];

  char* ws = (char*)d_ws;
  size_t off = 0;
  auto take = [&](size_t bytes) -> char* {
    char* p = ws + off;
    off += (bytes + 255) & ~(size_t)255;
    return p;
  };
  const long long NPIX = 4ll * 64 * 256 * 256;  // 16,777,216
  const long long NQK  = 4ll * 1024 * 1024;     //  4,194,304
  bf16_t* featbf = (bf16_t*)take(NPIX * 2);
  bf16_t* q_un   = (bf16_t*)take(NPIX * 2);
  bf16_t* k_un   = (bf16_t*)take(NPIX * 2);
  bf16_t* v_un   = (bf16_t*)take(NPIX * 2);
  bf16_t* wq_bf  = (bf16_t*)take(NQK * 2);
  bf16_t* wk_bf  = (bf16_t*)take(NQK * 2);
  bf16_t* wc_bf  = (bf16_t*)take(3ll * 36864 * 2);
  bf16_t* w1_bf  = (bf16_t*)take(16384ll * 2);
  bf16_t* w2_bf  = (bf16_t*)take(16384ll * 2);
  float*  q_lin  = (float*)take(NQK * 4);
  float*  k_lin  = (float*)take(NQK * 4);
  float*  qa     = (float*)take(1024 * 4);
  float*  qb     = (float*)take(1024 * 4);
  float*  ka     = (float*)take(1024 * 4);
  float*  kb     = (float*)take(1024 * 4);
  bf16_t* qp     = (bf16_t*)take(NQK * 2);
  bf16_t* kp     = (bf16_t*)take(NQK * 2);
  float*  scores = (float*)take(NQK * 4);
  bf16_t* attn   = (bf16_t*)take(NQK * 2);
  bf16_t* out_un = (bf16_t*)take(NPIX * 2);

  // dtype casts / weight packs
  cast_bf16_k<<<4096, 256, 0, stream>>>(feat, featbf, NPIX);
  cast_bf16_k<<<2048, 256, 0, stream>>>(wq, wq_bf, NQK);
  cast_bf16_k<<<2048, 256, 0, stream>>>(wk, wk_bf, NQK);
  pack_conv_w_k<<<64, 256, 0, stream>>>(w_enc, wc_bf);
  pack_conv_w_k<<<64, 256, 0, stream>>>(w_kc, wc_bf + 36864);
  pack_conv_w_k<<<64, 256, 0, stream>>>(w_vc, wc_bf + 2 * 36864);
  cast_bf16_k<<<64, 256, 0, stream>>>(w_f1, w1_bf, 16384ll);
  cast_bf16_k<<<64, 256, 0, stream>>>(w_f2, w2_bf, 16384ll);

  // 3 convs + GELU, writing unfolded q/k/v
  conv3x3_gelu_unfold_k<<<dim3(512, 4, 3), 256, 0, stream>>>(
      featbf, wc_bf, b_enc, b_kc, b_vc, q_un, k_un, v_un);

  // Q/K projections: [4096 x 4096] @ [4096 x 1024] (+bias) -> fp32
  gemm_bf16_k<true, 0><<<dim3(8, 32, 1), 256, 0, stream>>>(
      q_un, wq_bf, q_lin, bq, 4096, 1024, 4096, 0, 0, 0);
  gemm_bf16_k<true, 0><<<dim3(8, 32, 1), 256, 0, stream>>>(
      k_un, wk_bf, k_lin, bk, 4096, 1024, 4096, 0, 0, 0);

  // BN (batch stats) + fold 1/sqrt(1024) attention scale (1/sqrt(32) each side)
  const float ps = 0.17677669529663688f;
  bn_stats_k<<<1024, 256, 0, stream>>>(q_lin, gq, betaq, qa, qb, ps);
  bn_stats_k<<<1024, 256, 0, stream>>>(k_lin, gk, betak, ka, kb, ps);
  bn_apply_k<<<4096, 256, 0, stream>>>(q_lin, qa, qb, qp, NQK);
  bn_apply_k<<<4096, 256, 0, stream>>>(k_lin, ka, kb, kp, NQK);

  // scores[b] = qp[b] @ kp[b]^T  (kp row-major == [N][K] for B)
  gemm_bf16_k<true, 1><<<dim3(8, 8, 4), 256, 0, stream>>>(
      qp, kp, scores, nullptr, 1024, 1024, 1024, 1048576ll, 1048576ll,
      1048576ll);

  softmax_row_k<<<4096, 256, 0, stream>>>(scores, attn);

  // out[b] = attn[b] @ v_un[b]  (v is [K][N] -> transposed staging)
  gemm_bf16_k<false, 2><<<dim3(32, 8, 4), 256, 0, stream>>>(
      attn, v_un, out_un, nullptr, 1024, 4096, 1024, 1048576ll, 4194304ll,
      4194304ll);

  // fold + FFN + residual
  ffn_fused_k<<<dim3(1024, 4), 256, 0, stream>>>(
      out_un, w1_bf, b_f1, w2_bf, b_f2, feat, (float*)d_out);
}
